// MultiHeadAttention_42064909697126
// MI455X (gfx1250) — compile-verified, matrix operand order checked by reading, not run on hardware
//
#include <hip/hip_runtime.h>
#include <hip/hip_bf16.h>

// ---------------------------------------------------------------------------
// MHA forward for gfx1250 (CDNA5): bf16 WMMA everywhere, flash-style attention.
// B=4, S=2048, H=1024, NH=16, D=64. SCALE = 1/sqrt(64) = 0.125 (folded into Q).
// ---------------------------------------------------------------------------

#define B_  4
#define S_  2048
#define H_  1024
#define NH_ 16
#define D_  64
#define M_  (B_ * S_)          // 8192 rows for the projection GEMMs
#define SCALE_ 0.125f

typedef __attribute__((ext_vector_type(16))) __bf16 bf16x16;
typedef __attribute__((ext_vector_type(8)))  __bf16 bf16x8;
typedef __attribute__((ext_vector_type(8)))  float  f32x8;

// ---- helpers --------------------------------------------------------------

__device__ __forceinline__ __bf16 f2bf(float f) {
  union { float f; unsigned u; } v; v.f = f;
  unsigned r = v.u + 0x7FFFu + ((v.u >> 16) & 1u);   // round-to-nearest-even
  unsigned short h = (unsigned short)(r >> 16);
  return __builtin_bit_cast(__bf16, h);
}

// Load a 16x32 bf16 operand fragment (A operand, or B operand stored as
// [N,K] row-major which is layout-identical). Per ISA 7.12.2:
//   lanes 0..15  : row = lane,     K elems {0..7, 16..23}
//   lanes 16..31 : row = lane-16,  K elems {8..15, 24..31}
// ld must be a multiple of 8 elements; base 16B-aligned.
__device__ __forceinline__ bf16x16 load_frag(const __bf16* base, int ld) {
  const int lane = threadIdx.x & 31;
  const int row  = lane & 15;
  const int koff = (lane & 16) ? 8 : 0;
  const __bf16* p = base + row * ld + koff;
  bf16x8 lo = *(const bf16x8*)(p);
  bf16x8 hi = *(const bf16x8*)(p + 16);
  bf16x16 r;
#pragma unroll
  for (int j = 0; j < 8; ++j) { r[j] = lo[j]; r[j + 8] = hi[j]; }
  return r;
}

__device__ __forceinline__ f32x8 wmma_bf16(bf16x16 a, bf16x16 b, f32x8 c) {
  return __builtin_amdgcn_wmma_f32_16x16x32_bf16(
      /*neg_a=*/false, a, /*neg_b=*/false, b,
      /*c_mod=*/(short)0, c, /*reuse_a=*/false, /*reuse_b=*/false);
}

// ---- fp32 -> bf16 conversion ---------------------------------------------

__global__ void to_bf16_kernel(const float* __restrict__ in,
                               __bf16* __restrict__ out, int n) {
  int i = blockIdx.x * blockDim.x + threadIdx.x;
  if (i < n) out[i] = f2bf(in[i]);
}

// ---- fused QKV projection: y = x @ W^T + b --------------------------------
// Wave tile = 32(M) x 64(N): 2 A fragments, every B fragment reused across
// both row tiles -> 8 WMMAs per 12 b128 loads per K step.
// blockIdx.y selects {Q,K,V}. Outputs:
//   Q, K : bf16 [B, NH, S, D]   (Q pre-scaled by 1/sqrt(D))
//   V    : bf16 [B, NH, D, S]   (transposed for the P*V B-operand)
__global__ void qkv_gemm_kernel(const __bf16* __restrict__ xb,
                                const __bf16* __restrict__ wqb,
                                const __bf16* __restrict__ wkb,
                                const __bf16* __restrict__ wvb,
                                const float* __restrict__ bq,
                                const float* __restrict__ bk,
                                const float* __restrict__ bv,
                                __bf16* __restrict__ qout,
                                __bf16* __restrict__ kout,
                                __bf16* __restrict__ vtout) {
  const int wid   = (blockIdx.x * blockDim.x + threadIdx.x) >> 5;  // 0..4095
  const int mTile = wid >> 4;       // 256 tiles of 32 rows
  const int nGrp  = wid & 15;       // 16 groups of 64 cols
  const int z     = blockIdx.y;     // 0=Q 1=K 2=V
  const __bf16* W    = (z == 0) ? wqb : (z == 1) ? wkb : wvb;
  const float*  bias = (z == 0) ? bq  : (z == 1) ? bk  : bv;

  const int m0 = mTile * 32, n0 = nGrp * 64;
  f32x8 acc0[4] = {}, acc1[4] = {};

  for (int k = 0; k < H_; k += 32) {
    __builtin_prefetch(xb + (size_t)m0 * H_ + k + 128, 0, 1);   // global_prefetch_b8
    bf16x16 a0 = load_frag(xb + (size_t)m0 * H_ + k, H_);
    bf16x16 a1 = load_frag(xb + (size_t)(m0 + 16) * H_ + k, H_);
#pragma unroll
    for (int t = 0; t < 4; ++t) {
      bf16x16 b = load_frag(W + (size_t)(n0 + t * 16) * H_ + k, H_);
      acc0[t] = wmma_bf16(a0, b, acc0[t]);
      acc1[t] = wmma_bf16(a1, b, acc1[t]);
    }
  }

  const int lane = threadIdx.x & 31;
  const int colL = lane & 15;
  const int rAdd = (lane & 16) ? 8 : 0;
#pragma unroll
  for (int half = 0; half < 2; ++half) {
    const f32x8* acc = half ? acc1 : acc0;
    const int mbase = m0 + half * 16;
#pragma unroll
    for (int t = 0; t < 4; ++t) {
#pragma unroll
      for (int r = 0; r < 8; ++r) {
        const int m = mbase + r + rAdd;          // (b, s)
        const int n = n0 + t * 16 + colL;        // (h, d)
        const int bb = m >> 11, s = m & (S_ - 1);
        const int h  = n >> 6,  d = n & (D_ - 1);
        float v = acc[t][r] + bias[n];
        if (z == 0) v *= SCALE_;                 // fold 1/sqrt(D) into Q
        const __bf16 hv = f2bf(v);
        if (z == 2)      vtout[(((size_t)(bb * NH_ + h) * D_ + d) * S_) + s] = hv;
        else if (z == 1) kout [(((size_t)(bb * NH_ + h) * S_ + s) * D_) + d] = hv;
        else             qout [(((size_t)(bb * NH_ + h) * S_ + s) * D_) + d] = hv;
      }
    }
  }
}

// ---- flash attention ------------------------------------------------------
// Block = 128 threads (4 waves). Each wave owns a 32-query tile of one head:
// K and V fragments are loaded once per 32-key step and reused across both
// 16-query sub-tiles (16 WMMAs per 16 b128 loads). Online softmax; P staged
// through per-wave LDS to convert C-layout -> A-operand layout.
// ctx written bf16 as [B, S, H].
__global__ void attn_kernel(const __bf16* __restrict__ qb,
                            const __bf16* __restrict__ kb,
                            const __bf16* __restrict__ vtb,
                            __bf16* __restrict__ ctx) {
  __shared__ __bf16 pbuf[4][2][16 * 32];

  const int w    = threadIdx.x >> 5;
  const int lane = threadIdx.x & 31;
  const int bh   = blockIdx.x >> 4;         // 0..63 (b*NH + h)
  const int qblk = blockIdx.x & 15;         // 16 blocks of 128 queries
  const int bidx = bh >> 4, h = bh & 15;
  const int q0   = qblk * 128 + w * 32;     // query row within this head

  const __bf16* qp = qb  + ((size_t)bh * S_ + q0) * D_;
  const __bf16* kp = kb  + (size_t)bh * S_ * D_;
  const __bf16* vp = vtb + (size_t)bh * D_ * S_;

  bf16x16 qa[2][2];
  qa[0][0] = load_frag(qp, D_);
  qa[0][1] = load_frag(qp + 32, D_);
  qa[1][0] = load_frag(qp + 16 * D_, D_);
  qa[1][1] = load_frag(qp + 16 * D_ + 32, D_);

  f32x8 o[2][4] = {};
  float mrow[2][8], lrow[2][8];
#pragma unroll
  for (int qt = 0; qt < 2; ++qt)
#pragma unroll
    for (int r = 0; r < 8; ++r) { mrow[qt][r] = -3.0e38f; lrow[qt][r] = 0.0f; }

  const int prAdd = (lane & 16) ? 8 : 0;
  const int pcol  = lane & 15;

  for (int k0 = 0; k0 < S_; k0 += 32) {
    // K fragments for this 32-key block (shared by both query sub-tiles)
    bf16x16 k00 = load_frag(kp + (size_t)k0 * D_,             D_);
    bf16x16 k01 = load_frag(kp + (size_t)k0 * D_ + 32,        D_);
    bf16x16 k10 = load_frag(kp + (size_t)(k0 + 16) * D_,      D_);
    bf16x16 k11 = load_frag(kp + (size_t)(k0 + 16) * D_ + 32, D_);

#pragma unroll
    for (int qt = 0; qt < 2; ++qt) {
      f32x8 s0 = {}, s1 = {};
      s0 = wmma_bf16(qa[qt][0], k00, s0);  s0 = wmma_bf16(qa[qt][1], k01, s0);
      s1 = wmma_bf16(qa[qt][0], k10, s1);  s1 = wmma_bf16(qa[qt][1], k11, s1);

#pragma unroll
      for (int r = 0; r < 8; ++r) {
        float a0 = s0[r], a1 = s1[r];      // Q already carries 1/sqrt(D)
        // row max across the 16 lanes holding this row (xor<=8 stays in half)
        float mx = fmaxf(a0, a1);
        mx = fmaxf(mx, __shfl_xor(mx, 1));
        mx = fmaxf(mx, __shfl_xor(mx, 2));
        mx = fmaxf(mx, __shfl_xor(mx, 4));
        mx = fmaxf(mx, __shfl_xor(mx, 8));
        float mnew = fmaxf(mrow[qt][r], mx);
        float corr = __expf(mrow[qt][r] - mnew);
        float p0 = __expf(a0 - mnew);
        float p1 = __expf(a1 - mnew);
        float rs = p0 + p1;
        rs += __shfl_xor(rs, 1);
        rs += __shfl_xor(rs, 2);
        rs += __shfl_xor(rs, 4);
        rs += __shfl_xor(rs, 8);
        lrow[qt][r] = lrow[qt][r] * corr + rs;
        mrow[qt][r] = mnew;
        o[qt][0][r] *= corr; o[qt][1][r] *= corr;
        o[qt][2][r] *= corr; o[qt][3][r] *= corr;
        // stage P (16q x 32k) to LDS in plain row-major for A-frag reload
        const int prow = r + prAdd;
        pbuf[w][qt][prow * 32 + pcol]      = f2bf(p0);
        pbuf[w][qt][prow * 32 + pcol + 16] = f2bf(p1);
      }
    }
    __syncthreads();
    bf16x16 pa0 = load_frag(&pbuf[w][0][0], 32);   // ds_load_b128 path
    bf16x16 pa1 = load_frag(&pbuf[w][1][0], 32);
#pragma unroll
    for (int t = 0; t < 4; ++t) {
      bf16x16 vbf = load_frag(vp + (size_t)(t * 16) * S_ + k0, S_);
      o[0][t] = wmma_bf16(pa0, vbf, o[0][t]);
      o[1][t] = wmma_bf16(pa1, vbf, o[1][t]);
    }
    __syncthreads();
  }

  // normalize and emit ctx (bf16, [B, S, H])
#pragma unroll
  for (int qt = 0; qt < 2; ++qt) {
#pragma unroll
    for (int t = 0; t < 4; ++t) {
#pragma unroll
      for (int r = 0; r < 8; ++r) {
        const int row = r + prAdd;
        const int s   = q0 + qt * 16 + row;
        const int n   = h * D_ + t * 16 + pcol;
        ctx[((size_t)(bidx * S_ + s) * H_) + n] = f2bf(o[qt][t][r] / lrow[qt][r]);
      }
    }
  }
}

// ---- output projection: out = ctx @ Wo^T + bo (f32 result) ----------------

__global__ void out_gemm_kernel(const __bf16* __restrict__ ctx,
                                const __bf16* __restrict__ wob,
                                const float* __restrict__ bo,
                                float* __restrict__ out) {
  const int wid   = (blockIdx.x * blockDim.x + threadIdx.x) >> 5;  // 0..4095
  const int mTile = wid >> 4;       // 256 tiles of 32 rows
  const int nGrp  = wid & 15;
  const int m0 = mTile * 32, n0 = nGrp * 64;

  f32x8 acc0[4] = {}, acc1[4] = {};
  for (int k = 0; k < H_; k += 32) {
    __builtin_prefetch(ctx + (size_t)m0 * H_ + k + 128, 0, 1);
    bf16x16 a0 = load_frag(ctx + (size_t)m0 * H_ + k, H_);
    bf16x16 a1 = load_frag(ctx + (size_t)(m0 + 16) * H_ + k, H_);
#pragma unroll
    for (int t = 0; t < 4; ++t) {
      bf16x16 b = load_frag(wob + (size_t)(n0 + t * 16) * H_ + k, H_);
      acc0[t] = wmma_bf16(a0, b, acc0[t]);
      acc1[t] = wmma_bf16(a1, b, acc1[t]);
    }
  }

  const int lane = threadIdx.x & 31;
  const int colL = lane & 15;
  const int rAdd = (lane & 16) ? 8 : 0;
#pragma unroll
  for (int half = 0; half < 2; ++half) {
    const f32x8* acc = half ? acc1 : acc0;
    const int mbase = m0 + half * 16;
#pragma unroll
    for (int t = 0; t < 4; ++t) {
#pragma unroll
      for (int r = 0; r < 8; ++r) {
        const int m = mbase + r + rAdd;
        const int n = n0 + t * 16 + colL;
        out[(size_t)m * H_ + n] = acc[t][r] + bo[n];
      }
    }
  }
}

// ---- host side ------------------------------------------------------------

extern "C" void kernel_launch(void* const* d_in, const int* in_sizes, int n_in,
                              void* d_out, int out_size, void* d_ws, size_t ws_size,
                              hipStream_t stream) {
  (void)in_sizes; (void)n_in; (void)out_size; (void)ws_size;
  const float* x  = (const float*)d_in[0];
  const float* Wq = (const float*)d_in[1];
  const float* bq = (const float*)d_in[2];
  const float* Wk = (const float*)d_in[3];
  const float* bk = (const float*)d_in[4];
  const float* Wv = (const float*)d_in[5];
  const float* bv = (const float*)d_in[6];
  const float* Wo = (const float*)d_in[7];
  const float* bo = (const float*)d_in[8];
  float* out = (float*)d_out;

  char* ws = (char*)d_ws;
  size_t off = 0;
  auto carve = [&](size_t bytes) -> void* {
    void* p = ws + off;
    off = (off + bytes + 255) & ~(size_t)255;
    return p;
  };

  const size_t xElems = (size_t)M_ * H_;       // 8M
  const size_t wElems = (size_t)H_ * H_;       // 1M

  __bf16* xb    = (__bf16*)carve(xElems * 2);
  __bf16* wqb   = (__bf16*)carve(wElems * 2);
  __bf16* wkb   = (__bf16*)carve(wElems * 2);
  __bf16* wvb   = (__bf16*)carve(wElems * 2);
  __bf16* wob   = (__bf16*)carve(wElems * 2);
  __bf16* qbuf  = (__bf16*)carve(xElems * 2);  // [B,NH,S,D], pre-scaled
  __bf16* kbuf  = (__bf16*)carve(xElems * 2);  // [B,NH,S,D]
  __bf16* vtbuf = (__bf16*)carve(xElems * 2);  // [B,NH,D,S]
  __bf16* ctx   = (__bf16*)carve(xElems * 2);  // [B,S,H]

  // 1) fp32 -> bf16
  to_bf16_kernel<<<(int)((xElems + 255) / 256), 256, 0, stream>>>(x,  xb,  (int)xElems);
  to_bf16_kernel<<<(int)((wElems + 255) / 256), 256, 0, stream>>>(Wq, wqb, (int)wElems);
  to_bf16_kernel<<<(int)((wElems + 255) / 256), 256, 0, stream>>>(Wk, wkb, (int)wElems);
  to_bf16_kernel<<<(int)((wElems + 255) / 256), 256, 0, stream>>>(Wv, wvb, (int)wElems);
  to_bf16_kernel<<<(int)((wElems + 255) / 256), 256, 0, stream>>>(Wo, wob, (int)wElems);

  // 2) fused QKV projections: 4096 waves per projection -> 512 blocks x 8 waves
  qkv_gemm_kernel<<<dim3(512, 3), 256, 0, stream>>>(
      xb, wqb, wkb, wvb, bq, bk, bv, qbuf, kbuf, vtbuf);

  // 3) flash attention: B*NH*(S/128) = 1024 blocks, 4 waves each
  attn_kernel<<<1024, 128, 0, stream>>>(qbuf, kbuf, vtbuf, ctx);

  // 4) output projection (f32 result)
  out_gemm_kernel<<<512, 256, 0, stream>>>(ctx, wob, bo, out);
}
